// BertSelfAttention_36009005809934
// MI455X (gfx1250) — matrix operand order repaired and checked by reading, not compiled, not run on hardware
//
#include <hip/hip_runtime.h>
#include <hip/hip_bf16.h>

// ---------------------------------------------------------------------------
// BERT self-attention, fused: QKV projection (f16 WMMA GEMM) + flash attention
// B=8, S=1024, D=1024, H=16, DH=64.  Output f32 [B,S,D].
// Workspace: Q,K f16 [B,H,S,DH]; V f16 transposed [B,H,DH,S]  (48 MB total).
// Flash-attention K/V tile staging uses the Tensor Data Mover (TDM) with
// double buffering when the builtin is available; vector-load fallback kept.
// ---------------------------------------------------------------------------

typedef _Float16 v16h  __attribute__((ext_vector_type(16)));
typedef _Float16 half8 __attribute__((ext_vector_type(8)));
typedef _Float16 half4 __attribute__((ext_vector_type(4)));
typedef float    v8f   __attribute__((ext_vector_type(8)));

#ifndef __has_builtin
#define __has_builtin(x) 0
#endif
#if __has_builtin(__builtin_amdgcn_tensor_load_to_lds) && \
    __has_builtin(__builtin_amdgcn_s_wait_tensorcnt)
#define USE_TDM 1
#if __has_include(<hip/amd_detail/amd_gfx1250_TDM.h>)
#define TDM_6ARG 1   // therock-10.0 headers -> 6-arg builtin
#else
#define TDM_6ARG 0   // ROCm 7.2 -> 5-arg builtin
#endif
#else
#define USE_TDM 0
#endif

#define WMMA_F32_F16(A, B, C) \
  __builtin_amdgcn_wmma_f32_16x16x32_f16(false, (A), false, (B), (short)0, (C), false, false)

// A fragment (16x32 f16, M x K): lane L holds row m=L&15; halves 0..7 -> K=klo..klo+7,
// halves 8..15 -> K=16+klo..16+klo+7, klo = (L>=16)?8:0.   (ISA 7.12.2)
__device__ __forceinline__ v16h load_frag_A(const _Float16* base, int lane, int ld) {
  const int m   = lane & 15;
  const int klo = (lane & 16) >> 1;            // 0 or 8
  const _Float16* p = base + m * ld;
  half8 lo = *(const half8*)(p + klo);
  half8 hi = *(const half8*)(p + klo + 16);
  v16h r;
#pragma unroll
  for (int i = 0; i < 8; ++i) { r[i] = lo[i]; r[i + 8] = hi[i]; }
  return r;
}

// B fragment (32x16 f16, K x N) gathered from an LDS tile stored [n][k] (k contiguous):
// lane L holds col n=L&15, halves 0..7 -> K=koff..koff+7, halves 8..15 -> K=16+koff..16+koff+7, koff = (L>=16)?8:0.
__device__ __forceinline__ v16h load_frag_B(const _Float16* base, int lane, int ld) {
  const int n    = lane & 15;
  const int koff = (lane & 16) >> 1;           // 0 or 8
  const _Float16* p = base + n * ld + koff;
  half8 lo = *(const half8*)(p);
  half8 hi = *(const half8*)(p + 16);
  v16h r;
#pragma unroll
  for (int i = 0; i < 8; ++i) { r[i] = lo[i]; r[i + 8] = hi[i]; }
  return r;
}

#if USE_TDM
typedef unsigned int u32x4 __attribute__((ext_vector_type(4)));
typedef int          i32x4 __attribute__((ext_vector_type(4)));
typedef int          i32x8 __attribute__((ext_vector_type(8)));

// Generic LDS pointer: low 32 bits are the wave-relative LDS byte offset.
__device__ __forceinline__ unsigned lds_offset(const void* p) {
  return (unsigned)(unsigned long long)p;
}

// Issue a TDM 2-D tile load: rows x row_elems f16 elements, global row stride
// row_stride_elems, into LDS at lds_byte_off with +16B padding per 128B row
// (LDS row pitch = 144 B = 72 halves).
__device__ __forceinline__ void tdm_load_tile_2d(unsigned lds_byte_off,
                                                 const _Float16* gptr,
                                                 unsigned rows, unsigned row_elems,
                                                 unsigned row_stride_elems) {
  unsigned long long ga = (unsigned long long)(size_t)gptr;
  u32x4 g0;
  g0[0] = 1u;                                          // count=1 (valid), no gather
  g0[1] = lds_byte_off;                                // lds_addr
  g0[2] = (unsigned)(ga & 0xffffffffu);                // global_addr[31:0]
  g0[3] = (unsigned)((ga >> 32) & 0x1ffffffu) | (2u << 30);  // addr[56:32] | type=2
  i32x8 g1;
  // data_size=2B (1), pad_enable, pad_interval=32 DWORDs (code 4), pad_amount=4 DW (code 3)
  g1[0] = (int)((1u << 16) | (1u << 20) | (4u << 22) | (3u << 25));
  g1[1] = (int)(row_elems << 16);                      // tensor_dim0[15:0] (atomic addr = 0)
  g1[2] = (int)((row_elems >> 16) | (rows << 16));     // dim0[31:16] | tensor_dim1[15:0]
  g1[3] = (int)((rows >> 16) | (row_elems << 16));     // dim1[31:16] | tile_dim0
  g1[4] = (int)rows;                                   // tile_dim1 | tile_dim2=0
  g1[5] = (int)row_stride_elems;                       // tensor_dim0_stride[31:0]
  g1[6] = 0;                                           // stride hi | dim1_stride lo
  g1[7] = 0;
  i32x4 z4 = {0, 0, 0, 0};
#if TDM_6ARG
  i32x8 z8 = {0, 0, 0, 0, 0, 0, 0, 0};
  __builtin_amdgcn_tensor_load_to_lds(g0, g1, z4, z4, z8, 0);
#else
  __builtin_amdgcn_tensor_load_to_lds(g0, g1, z4, z4, 0);
#endif
}
#endif  // USE_TDM

// ---------------------------------------------------------------------------
// Kernel 1: C[8192,1024] = hidden @ W_e + b_e, stored f16 in attention layouts.
// grid = (8 col-tiles, 64 row-tiles, 3), block = 256 (8 waves).
// ---------------------------------------------------------------------------
__global__ __launch_bounds__(256) void qkv_proj_kernel(
    const float* __restrict__ hidden,
    const float* __restrict__ Wq, const float* __restrict__ bq,
    const float* __restrict__ Wk, const float* __restrict__ bk,
    const float* __restrict__ Wv, const float* __restrict__ bv,
    _Float16* __restrict__ qws, _Float16* __restrict__ kws, _Float16* __restrict__ vws) {
  const int e = blockIdx.z;
  const float* __restrict__ W    = (e == 0) ? Wq : ((e == 1) ? Wk : Wv);
  const float* __restrict__ bias = (e == 0) ? bq : ((e == 1) ? bk : bv);
  const int rowBase = blockIdx.y * 128;   // token index
  const int colBase = blockIdx.x * 128;   // output feature

  __shared__ __align__(16) _Float16 As[128 * 40];  // [row][k], pad 40 vs 32
  __shared__ __align__(16) _Float16 Bs[128 * 40];  // [n][k]

  const int tid  = threadIdx.x;
  const int lane = tid & 31;
  const int w    = tid >> 5;
  const int mrow = (w & 3) * 32;
  const int ncol = (w >> 2) * 64;

  v8f acc[2][4] = {};

  for (int k0 = 0; k0 < 1024; k0 += 32) {
    __syncthreads();
    // Stage A: 128 rows x 32 k (f32 -> f16)
#pragma unroll
    for (int it = 0; it < 4; ++it) {
      int job = tid + it * 256;
      int row = job >> 3;
      int c4  = (job & 7) * 4;
      float4 f = *(const float4*)(hidden + (size_t)(rowBase + row) * 1024 + k0 + c4);
      half4 hv;
      hv[0] = (_Float16)f.x; hv[1] = (_Float16)f.y;
      hv[2] = (_Float16)f.z; hv[3] = (_Float16)f.w;
      *(half4*)(&As[row * 40 + c4]) = hv;
    }
    // Stage B transposed: W[k][n] -> Bs[n][k]
#pragma unroll
    for (int it = 0; it < 4; ++it) {
      int job = tid + it * 256;
      int kr  = job >> 5;
      int c4  = (job & 31) * 4;
      float4 f = *(const float4*)(W + (size_t)(k0 + kr) * 1024 + colBase + c4);
      Bs[(c4 + 0) * 40 + kr] = (_Float16)f.x;
      Bs[(c4 + 1) * 40 + kr] = (_Float16)f.y;
      Bs[(c4 + 2) * 40 + kr] = (_Float16)f.z;
      Bs[(c4 + 3) * 40 + kr] = (_Float16)f.w;
    }
    __syncthreads();

    v16h a0 = load_frag_A(As + (size_t)mrow * 40, lane, 40);
    v16h a1 = load_frag_A(As + (size_t)(mrow + 16) * 40, lane, 40);
#pragma unroll
    for (int j = 0; j < 4; ++j) {
      v16h b = load_frag_B(Bs + (size_t)(ncol + j * 16) * 40, lane, 40);
      acc[0][j] = WMMA_F32_F16(a0, b, acc[0][j]);
      acc[1][j] = WMMA_F32_F16(a1, b, acc[1][j]);
    }
  }

  // Epilogue: bias add, f32 -> f16, scatter into attention layouts.
  const int mloc = (lane & 16) >> 1;  // 0 or 8
#pragma unroll
  for (int i = 0; i < 2; ++i) {
#pragma unroll
    for (int j = 0; j < 4; ++j) {
      int nglob = colBase + ncol + j * 16 + (lane & 15);
      float bvv = bias[nglob];
      int h = nglob >> 6, dh = nglob & 63;
#pragma unroll
      for (int r = 0; r < 8; ++r) {
        int t = rowBase + mrow + i * 16 + mloc + r;
        int b = t >> 10, s = t & 1023;
        _Float16 val = (_Float16)(acc[i][j][r] + bvv);
        if (e == 2) {
          vws[(((size_t)b * 16 + h) * 64 + dh) * 1024 + s] = val;      // [B,H,DH,S]
        } else {
          size_t idx = (((size_t)b * 16 + h) * 1024 + s) * 64 + dh;    // [B,H,S,DH]
          if (e == 0) qws[idx] = val; else kws[idx] = val;
        }
      }
    }
  }
}

// ---------------------------------------------------------------------------
// Kernel 2: flash attention per (b,h).  grid = (B*H=128, S/128=8), block 256.
// Each wave: 16 q-rows; K/V tiles (64 keys) double-buffered via TDM prefetch.
// ---------------------------------------------------------------------------
__global__ __launch_bounds__(256) void flash_attn_kernel(
    const _Float16* __restrict__ qws, const _Float16* __restrict__ kws,
    const _Float16* __restrict__ vws, const float* __restrict__ mask,
    float* __restrict__ out) {
  const int bh    = blockIdx.x;
  const int qBase = blockIdx.y * 128;
  const int b     = bh >> 4;
  const int h     = bh & 15;
  const int tid   = threadIdx.x;
  const int lane  = tid & 31;
  const int w     = tid >> 5;

  __shared__ __align__(16) _Float16 Qs[128 * 72];      // Q tile, reused as P buffer
  __shared__ __align__(16) _Float16 Ks[2][64 * 72];    // [key][dh], double-buffered
  __shared__ __align__(16) _Float16 Vs[2][64 * 72];    // [dh][key], double-buffered

  // Stage Q block [128 q, 64 dh]
#pragma unroll
  for (int it = 0; it < 4; ++it) {
    int job = tid + it * 256;
    int row = job >> 3, c8 = (job & 7) * 8;
    *(half8*)(&Qs[row * 72 + c8]) =
        *(const half8*)(qws + ((size_t)bh * 1024 + qBase + row) * 64 + c8);
  }
#if USE_TDM
  // Prefetch tile 0 into buffer 0 (TENSORcnt = 2 after this).
  if (w == 0) {
    tdm_load_tile_2d(lds_offset(&Ks[0][0]), kws + (size_t)bh * 1024 * 64, 64, 64, 64);
    tdm_load_tile_2d(lds_offset(&Vs[0][0]), vws + (size_t)bh * 64 * 1024, 64, 64, 1024);
  }
#endif
  __syncthreads();
  v16h qa0 = load_frag_A(Qs + (size_t)(w * 16) * 72, lane, 72);
  v16h qa1 = load_frag_A(Qs + (size_t)(w * 16) * 72 + 32, lane, 72);
  __syncthreads();  // Q consumed; Qs becomes the P buffer

  float mi[8], li[8];
#pragma unroll
  for (int r = 0; r < 8; ++r) { mi[r] = -1e30f; li[r] = 0.0f; }
  v8f oacc[4] = {};

  const float scale = 0.125f;  // 1/sqrt(64)
  _Float16* Pw = Qs + (size_t)(w * 16) * 72;
  const int mloc = (lane & 16) >> 1;

  for (int t = 0; t < 16; ++t) {
    const int kb  = t * 64;
    const int cur = t & 1;
#if USE_TDM
    if (w == 0) {
      if (t < 15) {
        const int nkb = kb + 64, nb = cur ^ 1;
        tdm_load_tile_2d(lds_offset(&Ks[nb][0]),
                         kws + ((size_t)bh * 1024 + nkb) * 64, 64, 64, 64);
        tdm_load_tile_2d(lds_offset(&Vs[nb][0]),
                         vws + (size_t)bh * 64 * 1024 + nkb, 64, 64, 1024);
        __builtin_amdgcn_s_wait_tensorcnt(2);  // tile t's two loads complete
      } else {
        __builtin_amdgcn_s_wait_tensorcnt(0);
      }
    }
#else
    // Fallback: vector staging of K [64 keys][64 dh] and Vt [64 dh][64 keys]
#pragma unroll
    for (int it = 0; it < 2; ++it) {
      int job = tid + it * 256;
      int row = job >> 3, c8 = (job & 7) * 8;
      *(half8*)(&Ks[cur][row * 72 + c8]) =
          *(const half8*)(kws + ((size_t)bh * 1024 + kb + row) * 64 + c8);
      *(half8*)(&Vs[cur][row * 72 + c8]) =
          *(const half8*)(vws + ((size_t)bh * 64 + row) * 1024 + kb + c8);
    }
#endif
    __syncthreads();

    // Scores: S[16q x 64key] = Q (16x64) @ K^T, 4 N-frags x 2 K-steps
    float sc[4][8];
#pragma unroll
    for (int j = 0; j < 4; ++j) {
      v16h bk0 = load_frag_B(Ks[cur] + (size_t)(j * 16) * 72, lane, 72);
      v16h bk1 = load_frag_B(Ks[cur] + (size_t)(j * 16) * 72 + 32, lane, 72);
      v8f s = {};
      s = WMMA_F32_F16(qa0, bk0, s);
      s = WMMA_F32_F16(qa1, bk1, s);
      float mk = mask[(size_t)b * 1024 + kb + j * 16 + (lane & 15)];
#pragma unroll
      for (int r = 0; r < 8; ++r) sc[j][r] = s[r] * scale + mk;
    }

    // Online softmax: row stats across 16 lanes in each half-wave.
    float rm[8];
#pragma unroll
    for (int r = 0; r < 8; ++r)
      rm[r] = fmaxf(fmaxf(sc[0][r], sc[1][r]), fmaxf(sc[2][r], sc[3][r]));
#pragma unroll
    for (int xm = 1; xm <= 8; xm <<= 1)
#pragma unroll
      for (int r = 0; r < 8; ++r) rm[r] = fmaxf(rm[r], __shfl_xor(rm[r], xm, 32));

    float al[8], rs[8];
#pragma unroll
    for (int r = 0; r < 8; ++r) {
      float mnew = fmaxf(mi[r], rm[r]);
      al[r] = __expf(mi[r] - mnew);
      mi[r] = mnew;
      rs[r] = 0.0f;
    }
#pragma unroll
    for (int j = 0; j < 4; ++j)
#pragma unroll
      for (int r = 0; r < 8; ++r) {
        float p = __expf(sc[j][r] - mi[r]);
        sc[j][r] = p;
        rs[r] += p;
      }
#pragma unroll
    for (int xm = 1; xm <= 8; xm <<= 1)
#pragma unroll
      for (int r = 0; r < 8; ++r) rs[r] += __shfl_xor(rs[r], xm, 32);
#pragma unroll
    for (int r = 0; r < 8; ++r) li[r] = li[r] * al[r] + rs[r];
#pragma unroll
    for (int jj = 0; jj < 4; ++jj)
#pragma unroll
      for (int r = 0; r < 8; ++r) oacc[jj][r] *= al[r];

    // P: C-frag layout -> LDS (per-wave region) -> A-frag layout
#pragma unroll
    for (int j = 0; j < 4; ++j)
#pragma unroll
      for (int r = 0; r < 8; ++r)
        Pw[(mloc + r) * 72 + j * 16 + (lane & 15)] = (_Float16)sc[j][r];
    __syncthreads();

    v16h pa0 = load_frag_A(Pw, lane, 72);
    v16h pa1 = load_frag_A(Pw + 32, lane, 72);
#pragma unroll
    for (int jj = 0; jj < 4; ++jj) {
      v16h bv0 = load_frag_B(Vs[cur] + (size_t)(jj * 16) * 72, lane, 72);
      v16h bv1 = load_frag_B(Vs[cur] + (size_t)(jj * 16) * 72 + 32, lane, 72);
      oacc[jj] = WMMA_F32_F16(pa0, bv0, oacc[jj]);
      oacc[jj] = WMMA_F32_F16(pa1, bv1, oacc[jj]);
    }
    __syncthreads();  // protect ping-pong buffers before next prefetch/stage
  }

  // Epilogue: O / l -> out[b, s, h*64+dh]  (f32)
  float rl[8];
#pragma unroll
  for (int r = 0; r < 8; ++r) rl[r] = 1.0f / li[r];
#pragma unroll
  for (int jj = 0; jj < 4; ++jj) {
    int dh = jj * 16 + (lane & 15);
#pragma unroll
    for (int r = 0; r < 8; ++r) {
      int s = qBase + w * 16 + mloc + r;
      out[((size_t)b * 1024 + s) * 1024 + h * 64 + dh] = oacc[jj][r] * rl[r];
    }
  }
}

// ---------------------------------------------------------------------------
extern "C" void kernel_launch(void* const* d_in, const int* in_sizes, int n_in,
                              void* d_out, int out_size, void* d_ws, size_t ws_size,
                              hipStream_t stream) {
  const float* hidden = (const float*)d_in[0];
  const float* mask   = (const float*)d_in[1];
  const float* Wq     = (const float*)d_in[2];
  const float* bq     = (const float*)d_in[3];
  const float* Wk     = (const float*)d_in[4];
  const float* bk     = (const float*)d_in[5];
  const float* Wv     = (const float*)d_in[6];
  const float* bv     = (const float*)d_in[7];
  float* out          = (float*)d_out;

  // f16 workspace: Q [B,H,S,DH], K [B,H,S,DH], V^T [B,H,DH,S] -> 3 * 16 MB
  const size_t qkv_elems = (size_t)8 * 16 * 1024 * 64;  // 8,388,608 halves each
  _Float16* qws = (_Float16*)d_ws;
  _Float16* kws = qws + qkv_elems;
  _Float16* vws = kws + qkv_elems;
  (void)in_sizes; (void)n_in; (void)out_size; (void)ws_size;

  dim3 g1(8, 64, 3), b1(256);
  qkv_proj_kernel<<<g1, b1, 0, stream>>>(hidden, Wq, bq, Wk, bk, Wv, bv, qws, kws, vws);

  dim3 g2(128, 8), b2(256);
  flash_attn_kernel<<<g2, b2, 0, stream>>>(qws, kws, vws, mask, out);
}
